// QuantBertSelfAttention_36739150250241
// MI455X (gfx1250) — 8-bit matrix operand order repaired and checked by reading, compiled, not run on hardware
//
#include <hip/hip_runtime.h>

#define SQ   2048
#define BATCH 2
#define HID  768
#define NH   12
#define DH   64
#define BS   (BATCH * SQ)
#define BHD  (BATCH * NH)
#define QB   127.0f

typedef __attribute__((ext_vector_type(8))) int v8i;

__device__ __forceinline__ int koffA(int j, int hi) {
    return (j & 1) * 4 + ((j >> 1) & 1) * 16 + (j >> 2) * 32 + hi * 8;
}
__device__ __forceinline__ int koffB(int j, int hi) {
    return (j & 1) * 4 + ((j >> 1) & 1) * 16 + (j >> 2) * 32 + hi * 8;
}

__device__ __forceinline__ signed char quant8(float v, float inv_scale) {
    float q = fminf(fmaxf(rintf(v * inv_scale), -QB), QB);
    return (signed char)(int)q;
}

__global__ void k_init(unsigned* sc) {
    sc[0] = 0u; sc[1] = 0u; sc[2] = 0u; sc[3] = 0u;
    sc[4] = 0x7F800000u;
}

__global__ void k_amax_x(const float* __restrict__ x, int n, unsigned* __restrict__ sc) {
    __shared__ float red[256];
    float m = 0.f;
    for (int i = blockIdx.x * blockDim.x + threadIdx.x; i < n; i += gridDim.x * blockDim.x)
        m = fmaxf(m, fabsf(x[i]));
    red[threadIdx.x] = m;
    __syncthreads();
    for (int s = 128; s > 0; s >>= 1) {
        if (threadIdx.x < (unsigned)s) red[threadIdx.x] = fmaxf(red[threadIdx.x], red[threadIdx.x + s]);
        __syncthreads();
    }
    if (threadIdx.x == 0) atomicMax(&sc[0], __float_as_uint(red[0]));
}

__global__ void k_amax_w(const float* __restrict__ W, float* __restrict__ aw) {
    __shared__ float red[128];
    const float* r = W + (size_t)blockIdx.x * HID;
    float m = 0.f;
    for (int i = threadIdx.x; i < HID; i += 128) m = fmaxf(m, fabsf(r[i]));
    red[threadIdx.x] = m;
    __syncthreads();
    for (int s = 64; s > 0; s >>= 1) {
        if (threadIdx.x < (unsigned)s) red[threadIdx.x] = fmaxf(red[threadIdx.x], red[threadIdx.x + s]);
        __syncthreads();
    }
    if (threadIdx.x == 0) aw[blockIdx.x] = red[0];
}

__global__ void k_quant_x(const float* __restrict__ x, signed char* __restrict__ xq,
                          const unsigned* __restrict__ sc, int n) {
    int i = blockIdx.x * blockDim.x + threadIdx.x;
    if (i >= n) return;
    float inv = QB / __uint_as_float(sc[0]);
    xq[i] = quant8(x[i], inv);
}

__global__ void k_quant_w(const float* __restrict__ W, signed char* __restrict__ wq,
                          const float* __restrict__ aw, int n) {
    int i = blockIdx.x * blockDim.x + threadIdx.x;
    if (i >= n) return;
    float inv = QB / aw[i / HID];
    wq[i] = quant8(W[i], inv);
}

__global__ void __launch_bounds__(128)
k_gemm_qkv(const signed char* __restrict__ Xq, const signed char* __restrict__ W8,
           const float* __restrict__ aw, unsigned* __restrict__ sc,
           const float* __restrict__ b0, const float* __restrict__ b1,
           const float* __restrict__ b2, float* __restrict__ outf) {
    int wave = blockIdx.x * 4 + (threadIdx.x >> 5);
    int lane = threadIdx.x & 31, lo = lane & 15, hi = lane >> 4;
    int mat = wave / 12288;
    int rem = wave % 12288;
    int stile = rem / 48, otile = rem % 48;

    const signed char* Wm = W8 + (size_t)mat * HID * HID;
    const float* bias = (mat == 0) ? b0 : ((mat == 1) ? b1 : b2);
    int srow = stile * 16 + lo;
    int ocol = otile * 16 + lo;

    v8i acc = {};
    for (int ch = 0; ch < 12; ++ch) {
        int kb = ch * 64;
        v8i a, b;
#pragma unroll
        for (int j = 0; j < 8; ++j) {
            a[j] = *(const int*)(Xq + (size_t)srow * HID + kb + koffA(j, hi));
            b[j] = *(const int*)(Wm + (size_t)ocol * HID + kb + koffB(j, hi));
        }
        acc = __builtin_amdgcn_wmma_i32_16x16x64_iu8(true, a, true, b, acc, false, false);
    }

    float sx = __uint_as_float(sc[0]) / QB;
    float sw = aw[mat * HID + ocol] / QB;
    float bb = bias[ocol];
    int head = ocol >> 6, d = ocol & 63;
    float* base = outf + (size_t)mat * BS * HID;
    float lmax = 0.f;
#pragma unroll
    for (int r = 0; r < 8; ++r) {
        int m = r + hi * 8;
        int sg = stile * 16 + m;
        int bidx = sg >> 11, sr = sg & 2047;
        float y = (float)acc[r] * sx * sw + bb;
        lmax = fmaxf(lmax, fabsf(y));
        base[(((size_t)(bidx * NH + head) * SQ + sr) * DH) + d] = y;
    }
    atomicMax(&sc[1 + mat], __float_as_uint(lmax));
}

__global__ void k_quant_qkv(const float* __restrict__ src, signed char* __restrict__ dst,
                            const unsigned* __restrict__ sc, int which, int transpose) {
    int i = blockIdx.x * blockDim.x + threadIdx.x;
    float inv = QB / __uint_as_float(sc[1 + which]);
    signed char qc = quant8(src[i], inv);
    if (!transpose) {
        dst[i] = qc;
    } else {
        int d  = i & 63;
        int t  = (i >> 6) & 2047;
        int bh = i >> 17;
        dst[((size_t)(bh * DH) + d) * SQ + t] = qc;
    }
}

__global__ void __launch_bounds__(128)
k_stats(const signed char* __restrict__ q8, const signed char* __restrict__ k8,
        const float* __restrict__ mask, const unsigned* __restrict__ sc,
        float* __restrict__ rowmax, float* __restrict__ rowsum) {
    __shared__ float scs[4][16][16];
    int wid = threadIdx.x >> 5, lane = threadIdx.x & 31;
    int lo = lane & 15, hi = lane >> 4;
    int wave = blockIdx.x * 4 + wid;
    int bh = wave >> 7, trow = wave & 127;
    int sbase = trow * 16;
    int b = bh / NH;

    float sscale = (__uint_as_float(sc[1]) / QB) * (__uint_as_float(sc[2]) / QB) * 0.125f;

    v8i a;
#pragma unroll
    for (int j = 0; j < 8; ++j)
        a[j] = *(const int*)(q8 + ((size_t)bh * SQ + sbase + lo) * DH + koffA(j, hi));

    float mrun = -3.0e38f, lrun = 0.f;
    for (int tc = 0; tc < 128; ++tc) {
        int t0 = tc * 16;
        v8i bfr;
#pragma unroll
        for (int j = 0; j < 8; ++j)
            bfr[j] = *(const int*)(k8 + ((size_t)bh * SQ + t0 + lo) * DH + koffB(j, hi));
        v8i z = {};
        v8i c = __builtin_amdgcn_wmma_i32_16x16x64_iu8(true, a, true, bfr, z, false, false);
        float mk = mask[b * SQ + t0 + lo];
#pragma unroll
        for (int r = 0; r < 8; ++r)
            scs[wid][r + hi * 8][lo] = (float)c[r] * sscale + mk;
        __syncthreads();
        if (lane < 16) {
            float cm = -3.0e38f;
#pragma unroll
            for (int n = 0; n < 16; ++n) cm = fmaxf(cm, scs[wid][lane][n]);
            float mnew = fmaxf(mrun, cm);
            float acc = lrun * __expf(mrun - mnew);
#pragma unroll
            for (int n = 0; n < 16; ++n) acc += __expf(scs[wid][lane][n] - mnew);
            mrun = mnew; lrun = acc;
        }
        __syncthreads();
    }
    if (lane < 16) {
        rowmax[bh * SQ + sbase + lane] = mrun;
        rowsum[bh * SQ + sbase + lane] = lrun;
    }
}

__global__ void k_minrow(const float* __restrict__ rowsum, int n, unsigned* __restrict__ sc) {
    __shared__ float red[256];
    float m = 3.0e38f;
    for (int i = blockIdx.x * blockDim.x + threadIdx.x; i < n; i += gridDim.x * blockDim.x)
        m = fminf(m, rowsum[i]);
    red[threadIdx.x] = m;
    __syncthreads();
    for (int s = 128; s > 0; s >>= 1) {
        if (threadIdx.x < (unsigned)s) red[threadIdx.x] = fminf(red[threadIdx.x], red[threadIdx.x + s]);
        __syncthreads();
    }
    if (threadIdx.x == 0) atomicMin(&sc[4], __float_as_uint(red[0]));
}

__global__ void __launch_bounds__(128)
k_attn(const signed char* __restrict__ q8, const signed char* __restrict__ k8,
       const signed char* __restrict__ v8T, const float* __restrict__ mask,
       const float* __restrict__ rowmax, const float* __restrict__ rowsum,
       const unsigned* __restrict__ sc, float* __restrict__ out) {
    __shared__ int ldsPI[4][256];
    int wid = threadIdx.x >> 5, lane = threadIdx.x & 31;
    int lo = lane & 15, hi = lane >> 4;
    int wave = blockIdx.x * 4 + wid;
    int bh = wave >> 7, trow = wave & 127;
    int sbase = trow * 16;
    int b = bh / NH, head = bh % NH;

    float sscale = (__uint_as_float(sc[1]) / QB) * (__uint_as_float(sc[2]) / QB) * 0.125f;
    float sv     = __uint_as_float(sc[3]) / QB;
    float ps     = (1.0f / __uint_as_float(sc[4])) / QB;
    float invps  = 1.0f / ps;

    v8i a;
#pragma unroll
    for (int j = 0; j < 8; ++j)
        a[j] = *(const int*)(q8 + ((size_t)bh * SQ + sbase + lo) * DH + koffA(j, hi));

    float rm[8], ri[8];
#pragma unroll
    for (int r = 0; r < 8; ++r) {
        int m = r + hi * 8;
        rm[r] = rowmax[bh * SQ + sbase + m];
        ri[r] = 1.0f / rowsum[bh * SQ + sbase + m];
    }

    v8i acc[4];
#pragma unroll
    for (int g = 0; g < 4; ++g) acc[g] = (v8i){};

    signed char* lp = (signed char*)ldsPI[wid];
    for (int tc = 0; tc < 32; ++tc) {
        int t0 = tc * 64;
#pragma unroll
        for (int f = 0; f < 4; ++f) {
            v8i bfr;
#pragma unroll
            for (int j = 0; j < 8; ++j)
                bfr[j] = *(const int*)(k8 + ((size_t)bh * SQ + t0 + f * 16 + lo) * DH + koffB(j, hi));
            v8i z = {};
            v8i c = __builtin_amdgcn_wmma_i32_16x16x64_iu8(true, a, true, bfr, z, false, false);
            float mk = mask[b * SQ + t0 + f * 16 + lo];
#pragma unroll
            for (int r = 0; r < 8; ++r) {
                float sval = (float)c[r] * sscale + mk;
                float p = __expf(sval - rm[r]) * ri[r];
                lp[(r + hi * 8) * 64 + f * 16 + lo] = quant8(p, invps);
            }
        }
        __syncthreads();
        v8i ap;
#pragma unroll
        for (int j = 0; j < 8; ++j)
            ap[j] = *(const int*)(lp + lo * 64 + koffA(j, hi));
#pragma unroll
        for (int g = 0; g < 4; ++g) {
            v8i bv;
#pragma unroll
            for (int j = 0; j < 8; ++j)
                bv[j] = *(const int*)(v8T + ((size_t)bh * DH + g * 16 + lo) * SQ + t0 + koffB(j, hi));
            acc[g] = __builtin_amdgcn_wmma_i32_16x16x64_iu8(true, ap, true, bv, acc[g], false, false);
        }
        __syncthreads();
    }

    float oscale = ps * sv;
#pragma unroll
    for (int g = 0; g < 4; ++g) {
        int d = g * 16 + lo;
#pragma unroll
        for (int r = 0; r < 8; ++r) {
            int m = r + hi * 8;
            int srow = sbase + m;
            out[((size_t)(b * SQ + srow)) * HID + head * DH + d] = (float)acc[g][r] * oscale;
        }
    }
}

extern "C" void kernel_launch(void* const* d_in, const int* in_sizes, int n_in,
                              void* d_out, int out_size, void* d_ws, size_t ws_size,
                              hipStream_t stream) {
    (void)in_sizes; (void)n_in; (void)out_size; (void)ws_size;
    const float* x    = (const float*)d_in[0];
    const float* mask = (const float*)d_in[1];
    const float* Wq   = (const float*)d_in[2];
    const float* bq   = (const float*)d_in[3];
    const float* Wk   = (const float*)d_in[4];
    const float* bk   = (const float*)d_in[5];
    const float* Wv   = (const float*)d_in[6];
    const float* bv   = (const float*)d_in[7];
    float* out = (float*)d_out;

    char* w = (char*)d_ws;
    size_t off = 0;
    auto alloc = [&](size_t bytes) -> void* {
        void* p = w + off;
        off = (off + bytes + 255) & ~(size_t)255;
        return p;
    };
    unsigned*    sc     = (unsigned*)alloc(8 * sizeof(unsigned));
    float*       aw     = (float*)alloc((size_t)3 * HID * sizeof(float));
    signed char* Xq     = (signed char*)alloc((size_t)BS * HID);
    signed char* W8     = (signed char*)alloc((size_t)3 * HID * HID);
    float*       qkvf   = (float*)alloc((size_t)3 * BS * HID * sizeof(float));
    signed char* q8     = (signed char*)alloc((size_t)BS * HID);
    signed char* k8     = (signed char*)alloc((size_t)BS * HID);
    signed char* v8T    = (signed char*)alloc((size_t)BS * HID);
    float*       rowmax = (float*)alloc((size_t)BHD * SQ * sizeof(float));
    float*       rowsum = (float*)alloc((size_t)BHD * SQ * sizeof(float));

    const int NX = BS * HID;
    const int NW = HID * HID;

    k_init<<<1, 1, 0, stream>>>(sc);
    k_amax_x<<<512, 256, 0, stream>>>(x, NX, sc);
    k_amax_w<<<HID, 128, 0, stream>>>(Wq, aw);
    k_amax_w<<<HID, 128, 0, stream>>>(Wk, aw + HID);
    k_amax_w<<<HID, 128, 0, stream>>>(Wv, aw + 2 * HID);
    k_quant_x<<<NX / 256, 256, 0, stream>>>(x, Xq, sc, NX);
    k_quant_w<<<NW / 256, 256, 0, stream>>>(Wq, W8, aw, NW);
    k_quant_w<<<NW / 256, 256, 0, stream>>>(Wk, W8 + (size_t)NW, aw + HID, NW);
    k_quant_w<<<NW / 256, 256, 0, stream>>>(Wv, W8 + (size_t)2 * NW, aw + 2 * HID, NW);
    k_gemm_qkv<<<9216, 128, 0, stream>>>(Xq, W8, aw, sc, bq, bk, bv, qkvf);
    k_quant_qkv<<<NX / 256, 256, 0, stream>>>(qkvf,                q8,  sc, 0, 0);
    k_quant_qkv<<<NX / 256, 256, 0, stream>>>(qkvf + (size_t)NX,   k8,  sc, 1, 0);
    k_quant_qkv<<<NX / 256, 256, 0, stream>>>(qkvf + (size_t)2*NX, v8T, sc, 2, 1);
    k_stats<<<768, 128, 0, stream>>>(q8, k8, mask, sc, rowmax, rowsum);
    k_minrow<<<48, 256, 0, stream>>>(rowsum, BHD * SQ, sc);
    k_attn<<<768, 128, 0, stream>>>(q8, k8, v8T, mask, rowmax, rowsum, sc, out);
}
